// LIFConv2dSTDP_72576357368391
// MI455X (gfx1250) — matrix operand order repaired and checked by reading, not compiled, not run on hardware
//
#include <hip/hip_runtime.h>
#include <hip/hip_bf16.h>

// ---------------- problem constants ----------------
#define T_STEPS 32
#define BATCH   8
#define C_IN    16
#define HH      64
#define WW      64
#define C_OUT   32
#define LPIX    (HH*WW)                 // 4096
#define NIN     (BATCH*C_IN*LPIX)       // 524288  (per-t input / pre-state)
#define NOUTT   (BATCH*C_OUT*LPIX)      // 1048576 (per-t output plane)
#define NOUT3   (T_STEPS*NOUTT)         // 33554432 per output tensor
#define KFEAT   144                     // C_IN*3*3
#define KPAD    160
#define NW      (C_OUT*KFEAT)           // 4608
#define HP      66                      // padded plane edge
#define PPLANE  (HP*HP)                 // 4356
#define NPAD    (BATCH*C_IN*PPLANE)     // 557568

#define BETA_M   0.9512294245007140f    // exp(-1/20)
#define BETA_S   0.8187307530779818f    // exp(-1/5)
#define BETA_PRE  BETA_M
#define BETA_POST BETA_M
#define V_TH    1.0f
#define T_REF   2
#define ETA_P   5.0e-4f
#define ETA_M   5.0e-4f
#define NORM    32768.0f                // B*L

typedef __attribute__((ext_vector_type(16))) __bf16 bf16x16;
typedef __attribute__((ext_vector_type(8)))  __bf16 bf16x8;
typedef __attribute__((ext_vector_type(8)))  float  floatx8;

// ---------------- workspace layout (float slots; all 32B aligned) ----------
#define WS_XPRE    0u           // f32 [524288]
#define WS_PRETR   524288u      // f32 [524288]
#define WS_POSTTR  1048576u     // f32 [1048576]
#define WS_REF     2097152u     // i32 [1048576]
#define WS_WT      3145728u     // f32 [4608]
#define WS_DWP     3150336u     // f32 [4608]
#define WS_DWM     3154944u     // f32 [4608]
#define WS_WBT     3159552u     // bf16[32*160]      (2560 slots)
#define WS_XPAD    3162112u     // bf16[NPAD]        (278784 slots)
#define WS_PPAD    3440896u     // bf16[NPAD]
#define WS_SPAD    3719680u     // bf16[NPAD]
#define WS_SPKBF   3998464u     // bf16[NOUTT]       (524288 slots)
#define WS_POSTBF  4522752u     // bf16[NOUTT]
// total 5047040 slots = 20.2 MB

// =================================================================
// K0: init state, zero padded halos, copy W0, pack bf16 weights
// =================================================================
__global__ void snn_init(const float* __restrict__ w0,
                         float* __restrict__ xpre, float* __restrict__ pretr,
                         float* __restrict__ posttr, int* __restrict__ refc,
                         float* __restrict__ wt, float* __restrict__ dwp,
                         float* __restrict__ dwm, __bf16* __restrict__ wbt,
                         __bf16* __restrict__ xpad, __bf16* __restrict__ ppad,
                         __bf16* __restrict__ spad) {
  unsigned idx = blockIdx.x * blockDim.x + threadIdx.x;
  if (idx < NOUTT) { posttr[idx] = 0.f; refc[idx] = 0; }
  if (idx < NPAD)  { xpad[idx] = (__bf16)0.0f; ppad[idx] = (__bf16)0.0f; spad[idx] = (__bf16)0.0f; }
  if (idx < NIN)   { xpre[idx] = 0.f; pretr[idx] = 0.f; }
  if (idx < NW)    { wt[idx] = w0[idx]; dwp[idx] = 0.f; dwm[idx] = 0.f; }
  if (idx < C_OUT * KPAD) {
    int n = idx / KPAD, kc = idx % KPAD;
    int s = kc >> 4, c = kc & 15;                  // conv K-order: k = s*16 + c
    wbt[idx] = (s < 9) ? (__bf16)w0[n * KFEAT + c * 9 + s] : (__bf16)0.0f;
  }
}

// =================================================================
// K1: synaptic + pre-trace update; refresh bf16 padded shadows
// =================================================================
__global__ void snn_trace(const float* __restrict__ s,
                          float* __restrict__ xpre, float* __restrict__ pretr,
                          __bf16* __restrict__ xpad, __bf16* __restrict__ ppad,
                          __bf16* __restrict__ spad) {
  unsigned idx = blockIdx.x * blockDim.x + threadIdx.x;
  if (idx >= NIN) return;
  int w = idx & 63, h = (idx >> 6) & 63;
  int bc = idx >> 12;                              // b*16 + c
  float sv = s[idx];
  float xn = xpre[idx] * BETA_S + sv;
  float pn = pretr[idx] * BETA_PRE + sv;
  xpre[idx] = xn;
  pretr[idx] = pn;
  int p = (bc * HP + (h + 1)) * HP + (w + 1);
  xpad[p] = (__bf16)xn;
  ppad[p] = (__bf16)pn;
  spad[p] = (__bf16)sv;
}

// =================================================================
// K2: implicit-GEMM 3x3 conv via v_wmma_f32_16x16x32_bf16
//     block = one (b,y) row, 4 waves (16-pixel M-tiles);
//     input slab staged in LDS as [dy][x][c] -> A halves are ds_load_b128
// =================================================================
__global__ void __launch_bounds__(128) snn_conv(const __bf16* __restrict__ xpad,
                                                const __bf16* __restrict__ wbt,
                                                float* __restrict__ iout) {
  __shared__ __bf16 tile[3 * HP * C_IN];           // 6336 B
  int b = blockIdx.x >> 6;
  int y = blockIdx.x & 63;

  // stage padded rows y..y+2, all 16 channels, transposed to channel-fastest
  for (int i = threadIdx.x; i < 3 * HP * C_IN; i += 128) {
    int dy = i / (HP * C_IN);
    int r  = i - dy * (HP * C_IN);
    int xx = r >> 4, c = r & 15;
    tile[i] = xpad[((b * C_IN + c) * HP + (y + dy)) * HP + xx];
  }
  __syncthreads();

  int xt = threadIdx.x >> 5;                       // wave id = M-tile
  int lane = threadIdx.x & 31;
  int m = lane & 15, half = lane >> 4;
  int x0 = xt * 16, x = x0 + m;

  __builtin_prefetch(wbt + m * KPAD, 0, 0);        // global_prefetch_b8

  bf16x8 hz;
#pragma unroll
  for (int j = 0; j < 8; ++j) hz[j] = (__bf16)0.0f;

  floatx8 c0, c1;
#pragma unroll
  for (int r = 0; r < 8; ++r) { c0[r] = 0.f; c1[r] = 0.f; }

#pragma unroll
  for (int i = 0; i < 5; ++i) {
    int s0 = 2 * i, s1 = s0 + 1;
    // tile offset for spatial tap s at pixel x, channel group cb:
    //   (( (s/3) * HP + (x + s%3) ) * 16 + cb)   -- 16B aligned
    int t0 = (((s0 / 3) * HP + (x + s0 % 3)) * C_IN) + half * 8;
    bf16x8 lo = *(const bf16x8*)(tile + t0);
    bf16x8 hi = hz;
    if (s1 < 9) {
      int t1 = (((s1 / 3) * HP + (x + s1 % 3)) * C_IN) + half * 8;
      hi = *(const bf16x8*)(tile + t1);
    }
    bf16x16 a = __builtin_shufflevector(lo, hi, 0, 1, 2, 3, 4, 5, 6, 7,
                                        8, 9, 10, 11, 12, 13, 14, 15);
    int k0h = 32 * i + half * 8;
    bf16x16 b0 = __builtin_shufflevector(*(const bf16x8*)(wbt + (lane & 15) * KPAD + k0h),
                                         *(const bf16x8*)(wbt + (lane & 15) * KPAD + k0h + 16),
                                         0, 1, 2, 3, 4, 5, 6, 7, 8, 9, 10, 11, 12, 13, 14, 15);
    bf16x16 b1 = __builtin_shufflevector(*(const bf16x8*)(wbt + ((lane & 15) + 16) * KPAD + k0h),
                                         *(const bf16x8*)(wbt + ((lane & 15) + 16) * KPAD + k0h + 16),
                                         0, 1, 2, 3, 4, 5, 6, 7, 8, 9, 10, 11, 12, 13, 14, 15);
    c0 = __builtin_amdgcn_wmma_f32_16x16x32_bf16(false, a, false, b0, (short)0, c0, false, false);
    c1 = __builtin_amdgcn_wmma_f32_16x16x32_bf16(false, a, false, b1, (short)0, c1, false, false);
  }

  int xs = x0 + half * 8;
  int n0 = lane & 15;
  *(floatx8*)(iout + ((b * C_OUT + n0)      * HH + y) * WW + xs) = c0;
  *(floatx8*)(iout + ((b * C_OUT + n0 + 16) * HH + y) * WW + xs) = c1;
}

// =================================================================
// K3: LIF dynamics + refractory + post-trace; bf16 shadows for STDP
// =================================================================
__global__ void snn_lif(const float* __restrict__ isyn, const float* __restrict__ vprev,
                        float* __restrict__ vout, float* __restrict__ spkout,
                        int* __restrict__ refc, float* __restrict__ posttr,
                        __bf16* __restrict__ spkbf, __bf16* __restrict__ postbf,
                        int first) {
  unsigned idx = blockIdx.x * blockDim.x + threadIdx.x;
  if (idx >= NOUTT) return;
  float vold = first ? 0.f : vprev[idx];
  float vv = vold * BETA_M + isyn[idx];
  int rc = refc[idx];
  bool refr = rc > 0;
  if (refr) vv = 0.f;
  bool spike = (vv >= V_TH) && !refr;
  if (spike) vv = 0.f;
  rc = spike ? T_REF : (refr ? rc - 1 : rc);
  if (rc < 0) rc = 0;
  float sf = spike ? 1.f : 0.f;
  float pt = posttr[idx] * BETA_POST + sf;
  posttr[idx] = pt;
  refc[idx] = rc;
  vout[idx] = vv;
  spkout[idx] = sf;
  spkbf[idx] = (__bf16)sf;
  postbf[idx] = (__bf16)pt;
}

// =================================================================
// K4: STDP correlations: dWp += Ys·Xp^T, dWm += Yp·Xs^T (WMMA bf16)
//     block = 2 waves (o-tiles); grid = 9 n-tiles x 64 contraction slices
//     all fragment loads unconditional (padded/bf16 shadows)
// =================================================================
__global__ void __launch_bounds__(64) snn_stdp(const __bf16* __restrict__ ppad,
                                               const __bf16* __restrict__ spad,
                                               const __bf16* __restrict__ spkbf,
                                               const __bf16* __restrict__ postbf,
                                               float* __restrict__ dwp,
                                               float* __restrict__ dwm) {
  int lane = threadIdx.x & 31;
  int otile = threadIdx.x >> 5;                    // 0..1
  int ntile = blockIdx.x / 64;                     // 0..8
  int slice = blockIdx.x % 64;                     // 0..63
  int m = lane & 15, half = lane >> 4;
  int kk = ntile * 16 + m;                         // 0..143 (unfold order c*9+s)
  int cch = kk / 9, s = kk % 9;
  int dy = s / 3 - 1, dx = s % 3 - 1;
  int o = otile * 16 + m;

  floatx8 accP, accM;
#pragma unroll
  for (int r = 0; r < 8; ++r) { accP[r] = 0.f; accM[r] = 0.f; }

  for (int ci = 0; ci < 16; ++ci) {
    int u0 = (slice * 16 + ci) * 32;
    int b = u0 >> 12, l0 = u0 & 4095;

    // --- A fragments (Y: spikes / post-trace) -- aligned 16B bf16 loads
    const __bf16* ysrow = spkbf  + (b * C_OUT + o) * LPIX + l0 + half * 8;
    const __bf16* yprow = postbf + (b * C_OUT + o) * LPIX + l0 + half * 8;
    bf16x8 sLo = *(const bf16x8*)(ysrow);
    bf16x8 sHi = *(const bf16x8*)(ysrow + 16);
    bf16x8 pLo = *(const bf16x8*)(yprow);
    bf16x8 pHi = *(const bf16x8*)(yprow + 16);
    bf16x16 aS = __builtin_shufflevector(sLo, sHi, 0, 1, 2, 3, 4, 5, 6, 7,
                                         8, 9, 10, 11, 12, 13, 14, 15);
    bf16x16 aP = __builtin_shufflevector(pLo, pHi, 0, 1, 2, 3, 4, 5, 6, 7,
                                         8, 9, 10, 11, 12, 13, 14, 15);

    // --- B fragments (X patches) -- unconditional contiguous u16 loads
    int y = l0 >> 6;
    int xb = (l0 & 63) + half * 8;
    int base = (((b * C_IN + cch) * HP) + (y + dy + 1)) * HP + (xb + dx + 1);
    bf16x16 bP, bS;
#pragma unroll
    for (int j = 0; j < 16; ++j) {
      bP[j] = ppad[base + j + (j & 8)];
      bS[j] = spad[base + j + (j & 8)];
    }

    accP = __builtin_amdgcn_wmma_f32_16x16x32_bf16(false, aS, false, bP, (short)0, accP, false, false);
    accM = __builtin_amdgcn_wmma_f32_16x16x32_bf16(false, aP, false, bS, (short)0, accM, false, false);
  }

#pragma unroll
  for (int r = 0; r < 8; ++r) {
    int oo = otile * 16 + half * 8 + r;            // D layout: VGPR r -> M = r (+8 hi lanes)
    int j = oo * KFEAT + kk;
    atomicAdd(&dwp[j], accP[r]);
    atomicAdd(&dwm[j], accM[r]);
  }
}

// =================================================================
// K5: apply soft-bound STDP update, clip, re-zero dW, repack bf16 W
// =================================================================
__global__ void snn_wupd(float* __restrict__ wt, float* __restrict__ dwp,
                         float* __restrict__ dwm, __bf16* __restrict__ wbt) {
  unsigned idx = blockIdx.x * blockDim.x + threadIdx.x;
  if (idx >= C_OUT * KPAD) return;
  int n = idx / KPAD, kc = idx % KPAD;
  int s = kc >> 4, c = kc & 15;                    // conv K-order k = s*16 + c
  if (s < 9) {
    int j = n * KFEAT + c * 9 + s;
    float w = wt[j];
    w += (ETA_P / NORM) * dwp[j] * (1.0f - w);
    w -= (ETA_M / NORM) * dwm[j] * (w - 0.0f);
    w = fminf(fmaxf(w, 0.0f), 1.0f);
    wt[j] = w;
    dwp[j] = 0.f;
    dwm[j] = 0.f;
    wbt[idx] = (__bf16)w;
  } else {
    wbt[idx] = (__bf16)0.0f;
  }
}

// =================================================================
extern "C" void kernel_launch(void* const* d_in, const int* in_sizes, int n_in,
                              void* d_out, int out_size, void* d_ws, size_t ws_size,
                              hipStream_t stream) {
  const float* S  = (const float*)d_in[0];         // (T,B,C_IN,H,W)
  const float* W0 = (const float*)d_in[1];         // (C_OUT,C_IN,3,3)

  float* out    = (float*)d_out;
  float* spikes = out;                             // (T,B,C_OUT,H,W)
  float* vtraj  = out + (size_t)NOUT3;
  float* itraj  = out + (size_t)2 * NOUT3;

  float* ws      = (float*)d_ws;
  float* xpre    = ws + WS_XPRE;
  float* pretr   = ws + WS_PRETR;
  float* posttr  = ws + WS_POSTTR;
  int*   refc    = (int*)(ws + WS_REF);
  float* wt      = ws + WS_WT;
  float* dwp     = ws + WS_DWP;
  float* dwm     = ws + WS_DWM;
  __bf16* wbt    = (__bf16*)(ws + WS_WBT);
  __bf16* xpad   = (__bf16*)(ws + WS_XPAD);
  __bf16* ppad   = (__bf16*)(ws + WS_PPAD);
  __bf16* spad   = (__bf16*)(ws + WS_SPAD);
  __bf16* spkbf  = (__bf16*)(ws + WS_SPKBF);
  __bf16* postbf = (__bf16*)(ws + WS_POSTBF);

  snn_init<<<(NOUTT + 255) / 256, 256, 0, stream>>>(W0, xpre, pretr, posttr, refc,
                                                    wt, dwp, dwm, wbt, xpad, ppad, spad);

  for (int t = 0; t < T_STEPS; ++t) {
    const float* st = S + (size_t)t * NIN;
    float* it = itraj + (size_t)t * NOUTT;
    float* vt = vtraj + (size_t)t * NOUTT;
    float* vp = (t == 0) ? vtraj : (vtraj + (size_t)(t - 1) * NOUTT);
    float* sp = spikes + (size_t)t * NOUTT;

    snn_trace<<<(NIN + 255) / 256, 256, 0, stream>>>(st, xpre, pretr, xpad, ppad, spad);
    snn_conv <<<512, 128, 0, stream>>>(xpad, wbt, it);
    snn_lif  <<<(NOUTT + 255) / 256, 256, 0, stream>>>(it, vp, vt, sp, refc, posttr,
                                                       spkbf, postbf, (t == 0) ? 1 : 0);
    snn_stdp <<<9 * 64, 64, 0, stream>>>(ppad, spad, spkbf, postbf, dwp, dwm);
    snn_wupd <<<(C_OUT * KPAD + 255) / 256, 256, 0, stream>>>(wt, dwp, dwm, wbt);
  }
}
